// HiResPrecipNet_CNN_GNN_new_1563368096261
// MI455X (gfx1250) — compile-verified, matrix operand order checked by reading, not run on hardware
//
#include <hip/hip_runtime.h>
#include <cstdint>
#include <cstddef>

// ---------------------------------------------------------------------------
// Types for CDNA5 WMMA (wave32, 16x16x32 bf16 -> f32)
// ---------------------------------------------------------------------------
typedef __attribute__((ext_vector_type(16))) __bf16 v16bf;
typedef __attribute__((ext_vector_type(8)))  __bf16 v8bf;
typedef __attribute__((ext_vector_type(8)))  float  v8f;

#define WG 256

// ---------------------------------------------------------------------------
// helpers
// ---------------------------------------------------------------------------
static __device__ __forceinline__ unsigned f2ord(float f) {
  unsigned b = __float_as_uint(f);
  return (b & 0x80000000u) ? ~b : (b | 0x80000000u);
}
static __device__ __forceinline__ float ord2f(unsigned u) {
  unsigned b = (u & 0x80000000u) ? (u & 0x7fffffffu) : ~u;
  return __uint_as_float(b);
}
static __device__ __forceinline__ float lrelu(float x) { return x > 0.f ? x : 0.2f * x; }
static __device__ __forceinline__ float relu_(float x) { return x > 0.f ? x : 0.f; }

// ---------------------------------------------------------------------------
// generic fills
// ---------------------------------------------------------------------------
__global__ void fill_f32(float* p, float v, long n) {
  long i = (long)blockIdx.x * blockDim.x + threadIdx.x;
  long stride = (long)gridDim.x * blockDim.x;
  for (; i < n; i += stride) p[i] = v;
}
__global__ void fill_u32(unsigned* p, unsigned v, long n) {
  long i = (long)blockIdx.x * blockDim.x + threadIdx.x;
  long stride = (long)gridDim.x * blockDim.x;
  for (; i < n; i += stride) p[i] = v;
}

// ---------------------------------------------------------------------------
// CNN encoder: one block per low-res node, entire stack fused in LDS.
// input x_low node: (C=1, D=5, H=25, W=5); output enc_low: 32 features.
// ---------------------------------------------------------------------------
struct CnnParams {
  const float* tw[5]; const float* tb[5];
  const float* tg[5]; const float* tbe[5]; const float* tm[5]; const float* tv[5];
  const float* vw[2]; const float* vb[2];
  const float* vg[2]; const float* vbe[2]; const float* vm[2]; const float* vv[2];
  const float* dW;    const float* db;
};

static __device__ void bn_fold(const float* g, const float* be, const float* mn,
                               const float* vr, const float* b, int C,
                               float* s, float* t, int tid) {
  for (int c = tid; c < C; c += WG) {
    float sc = g[c] * rsqrtf(vr[c] + 1e-5f);
    s[c] = sc;
    t[c] = (b[c] - mn[c]) * sc + be[c];
  }
}

// conv along H, kernel 5, no pad, fused bn+relu. layout ((c*5+d)*H+h)*5+w
static __device__ void conv_h5(const float* in, float* out, const float* w,
                               int Cin, int Cout, int Hin,
                               const float* s, const float* t, int tid) {
  int Hout = Hin - 4;
  int total = Cout * 5 * Hout * 5;
  for (int i = tid; i < total; i += WG) {
    int ww = i % 5; int tmp = i / 5;
    int ho = tmp % Hout; tmp /= Hout;
    int dd = tmp % 5; int co = tmp / 5;
    float acc = 0.f;
    for (int ci = 0; ci < Cin; ++ci) {
      const float* ip = in + ((size_t)(ci * 5 + dd) * Hin + ho) * 5 + ww;
      const float* wp = w + (size_t)(co * Cin + ci) * 5;
      #pragma unroll
      for (int kh = 0; kh < 5; ++kh) acc += ip[kh * 5] * wp[kh];
    }
    float v = acc * s[co] + t[co];
    out[((size_t)(co * 5 + dd) * Hout + ho) * 5 + ww] = relu_(v);
  }
}

// conv along W, kernel 3, pad 0/1, fused bn+relu. layout (c*5+d)*W+w (H==1)
static __device__ void conv_w3(const float* in, float* out, const float* w,
                               int Cin, int Cout, int Win, int pad,
                               const float* s, const float* t, int tid) {
  int Wout = Win + 2 * pad - 2;
  int total = Cout * 5 * Wout;
  for (int i = tid; i < total; i += WG) {
    int wo = i % Wout; int tmp = i / Wout;
    int dd = tmp % 5; int co = tmp / 5;
    float acc = 0.f;
    for (int ci = 0; ci < Cin; ++ci) {
      const float* ip = in + (size_t)(ci * 5 + dd) * Win;
      const float* wp = w + (size_t)(co * Cin + ci) * 3;
      #pragma unroll
      for (int kw = 0; kw < 3; ++kw) {
        int wi = wo - pad + kw;
        if (wi >= 0 && wi < Win) acc += ip[wi] * wp[kw];
      }
    }
    float v = acc * s[co] + t[co];
    out[(size_t)(co * 5 + dd) * Wout + wo] = relu_(v);
  }
}

__global__ __launch_bounds__(WG)
void cnn_encoder(const float* __restrict__ x_low, CnnParams cp,
                 float* __restrict__ enc, int Nl) {
  __shared__ float smem[15808];   // bufIn 625 | bufA 8320 | bufB 6800
  __shared__ float sS[32], sT[32];
  float* bufIn = smem;
  float* bufA  = smem + 640;
  float* bufB  = smem + 640 + 8320;
  int tid = threadIdx.x;
  int node = blockIdx.x;
  if (node >= Nl) return;

  const int tc[6] = {1, 8, 16, 32, 16, 8};
  const int hs[6] = {25, 21, 17, 13, 9, 5};

  // load input
  const float* xin = x_low + (size_t)node * 625;
  for (int i = tid; i < 625; i += WG) bufIn[i] = xin[i];
  __syncthreads();

  // 5 temporal convs (kernel (1,5,1))
  const float* src = bufIn;
  float* bufs[2] = {bufA, bufB};
  for (int l = 0; l < 5; ++l) {
    bn_fold(cp.tg[l], cp.tbe[l], cp.tm[l], cp.tv[l], cp.tb[l], tc[l + 1], sS, sT, tid);
    __syncthreads();
    float* dst = bufs[l & 1];
    conv_h5(src, dst, cp.tw[l], tc[l], tc[l + 1], hs[l], sS, sT, tid);
    __syncthreads();
    src = dst;
  }
  // maxpool (1,5,1) stride (1,2,1): (8,5,5,5) -> (8,5,1,5)   [src == bufA]
  for (int i = tid; i < 8 * 5 * 5; i += WG) {
    int w = i % 5; int tmp = i / 5; int d = tmp % 5; int c = tmp / 5;
    float m = -3.4e38f;
    #pragma unroll
    for (int h = 0; h < 5; ++h) {
      float v = src[(((size_t)(c * 5 + d)) * 5 + h) * 5 + w];
      m = fmaxf(m, v);
    }
    bufB[(size_t)(c * 5 + d) * 5 + w] = m;
  }
  __syncthreads();
  // vconv0 (16,8,1,1,3) pad W=1: (8,5,5) -> (16,5,5)
  bn_fold(cp.vg[0], cp.vbe[0], cp.vm[0], cp.vv[0], cp.vb[0], 16, sS, sT, tid);
  __syncthreads();
  conv_w3(bufB, bufA, cp.vw[0], 8, 16, 5, 1, sS, sT, tid);
  __syncthreads();
  // vconv1 (32,16,1,1,3) pad 0: (16,5,5) -> (32,5,3)
  bn_fold(cp.vg[1], cp.vbe[1], cp.vm[1], cp.vv[1], cp.vb[1], 32, sS, sT, tid);
  __syncthreads();
  conv_w3(bufA, bufB, cp.vw[1], 16, 32, 5, 0, sS, sT, tid);
  __syncthreads();
  // maxpool (1,1,3) stride (1,1,2): (32,5,3) -> (32,5,1) -> flat 160
  for (int i = tid; i < 160; i += WG) {
    int d = i % 5; int c = i / 5;
    float m = -3.4e38f;
    #pragma unroll
    for (int w = 0; w < 3; ++w) m = fmaxf(m, bufB[(size_t)(c * 5 + d) * 3 + w]);
    bufA[i] = m;  // idx = c*5+d matches reshape order
  }
  __syncthreads();
  // dense 160 -> 32 + relu
  for (int o = tid; o < 32; o += WG) {
    float acc = cp.db[o];
    for (int i = 0; i < 160; ++i) acc += bufA[i] * cp.dW[(size_t)i * 32 + o];
    enc[(size_t)node * 32 + o] = relu_(acc);
  }
}

// ---------------------------------------------------------------------------
// bf16 WMMA GEMM:  Y[M,N] = A[M,K] @ W[K,N] + bias  (optional relu)
// Block = 8 waves; each wave owns a 16x64 output strip (4 n-tiles).
//
// Tiles are staged into LDS *pre-permuted into WMMA fragment order*:
// per-lane data is contiguous, so a fragment is a single 32-byte LDS read
// (2x ds_load_b128) instead of 16 scalar ds_load_u16 + v_mov_b16 packs.
//
// Fragment layout (CDNA5 ISA §7.12.2, 16-bit A 16x32 / B 32x16):
//   lane = half*16 + (row|col),  elem j <-> K = (j<8 ? j : j+8) + half*8
// Inverse used at staging time: for K-octet ko in {0,8,16,24}:
//   half = (ko>>3)&1,  jbase = (ko&7) + (ko>=16 ? 8 : 0)
// and the 8 consecutive K values fill 8 consecutive j slots of one lane.
// ---------------------------------------------------------------------------
#define TM 128
#define TN 64
#define TK 32

__global__ __launch_bounds__(WG)
void gemm_bf16_wmma(const float* __restrict__ A, const float* __restrict__ W,
                    const float* __restrict__ bias, float* __restrict__ Y,
                    int M, int K, int N, int relu_flag) {
  // fragment-ordered tiles: [warp-or-ntile][lane][16 elems]
  __shared__ __align__(32) __bf16 AldsF[(TM / 16) * 32 * 16];  // 4096 = 8KB
  __shared__ __align__(32) __bf16 BldsF[(TN / 16) * 32 * 16];  // 2048 = 4KB
  int tid  = threadIdx.x;
  int wave = tid >> 5;
  int lane = tid & 31;
  int half = lane >> 4;     // lane group (0: lanes 0-15, 1: lanes 16-31)
  int lr   = lane & 15;     // row (A) / col (B,C) within tile
  int m0 = blockIdx.x * TM;
  int n0 = blockIdx.y * TN;

  v8f acc[4];
  #pragma unroll
  for (int nt = 0; nt < 4; ++nt) acc[nt] = v8f{0.f,0.f,0.f,0.f,0.f,0.f,0.f,0.f};

  int ksteps = (K + TK - 1) / TK;
  for (int ks = 0; ks < ksteps; ++ks) {
    int k0 = ks * TK;
    __syncthreads();
    // stage A tile (128 rows x 32 K) -> fragment order. 512 octet tasks.
    for (int task = tid; task < TM * (TK / 8); task += WG) {
      int r  = task >> 2;             // 0..127
      int ko = (task & 3) * 8;        // 0,8,16,24
      int gm = m0 + r;
      int wv = r >> 4, rr = r & 15;
      int hf = (ko >> 3) & 1;
      int jb = (ko & 7) + (ko >= 16 ? 8 : 0);
      int ln = hf * 16 + rr;
      v8bf pk;
      #pragma unroll
      for (int u = 0; u < 8; ++u) {
        int gk = k0 + ko + u;
        float v = (gm < M && gk < K) ? A[(size_t)gm * K + gk] : 0.f;
        pk[u] = (__bf16)v;
      }
      *(v8bf*)&AldsF[(size_t)(wv * 32 + ln) * 16 + jb] = pk;
    }
    // stage B tile (32 K x 64 n) -> fragment order. 256 octet tasks.
    for (int task = tid; task < TN * (TK / 8); task += WG) {
      int n  = task & 63;             // col within tile
      int ko = (task >> 6) * 8;
      int gn = n0 + n;
      int nt = n >> 4, cc = n & 15;
      int hf = (ko >> 3) & 1;
      int jb = (ko & 7) + (ko >= 16 ? 8 : 0);
      int ln = hf * 16 + cc;
      v8bf pk;
      #pragma unroll
      for (int u = 0; u < 8; ++u) {
        int gk = k0 + ko + u;
        float v = (gk < K && gn < N) ? W[(size_t)gk * N + gn] : 0.f;
        pk[u] = (__bf16)v;
      }
      *(v8bf*)&BldsF[(size_t)(nt * 32 + ln) * 16 + jb] = pk;
    }
    __syncthreads();

    // fragments are now single contiguous 32B per-lane reads
    v16bf af = *(const v16bf*)&AldsF[(size_t)(wave * 32 + lane) * 16];
    #pragma unroll
    for (int nt = 0; nt < 4; ++nt) {
      v16bf bfr = *(const v16bf*)&BldsF[(size_t)(nt * 32 + lane) * 16];
      acc[nt] = __builtin_amdgcn_wmma_f32_16x16x32_bf16(
          false, af, false, bfr, (short)0, acc[nt], false, false);
    }
  }

  // store C (f32 16x16 layout: elem r -> row r + half*8, col = lr)
  #pragma unroll
  for (int nt = 0; nt < 4; ++nt) {
    int col = n0 + nt * 16 + lr;
    if (col >= N) continue;
    float bv = bias ? bias[col] : 0.f;
    #pragma unroll
    for (int r = 0; r < 8; ++r) {
      int row = m0 + wave * 16 + r + half * 8;
      if (row < M) {
        float v = acc[nt][r] + bv;
        if (relu_flag) v = relu_(v);
        Y[(size_t)row * N + col] = v;
      }
    }
  }
}

// ---------------------------------------------------------------------------
// GAT edge kernels (generic heads H, C=64)
// ---------------------------------------------------------------------------
__global__ void edge_alpha(const float* __restrict__ xl, const float* __restrict__ xr,
                           const int* __restrict__ src, const int* __restrict__ dst,
                           const float* __restrict__ att, float* __restrict__ alpha,
                           unsigned* __restrict__ amax, int E, int H, int C) {
  int t = blockIdx.x * blockDim.x + threadIdx.x;
  if (t >= E * H) return;
  int e = t / H, h = t - e * H;
  int s = src[e], d = dst[e];
  const float* pl = xl + (size_t)s * H * C + (size_t)h * C;
  const float* pr = xr + (size_t)d * H * C + (size_t)h * C;
  const float* pa = att + (size_t)h * C;
  float acc = 0.f;
  for (int c = 0; c < C; ++c) acc += lrelu(pl[c] + pr[c]) * pa[c];
  alpha[t] = acc;
  atomicMax(&amax[(size_t)d * H + h], f2ord(acc));
}

__global__ void edge_exp(const float* __restrict__ alpha, const unsigned* __restrict__ amax,
                         const int* __restrict__ dst, float* __restrict__ ea,
                         float* __restrict__ den, float* __restrict__ deg,
                         int E, int H) {
  int t = blockIdx.x * blockDim.x + threadIdx.x;
  if (t >= E * H) return;
  int e = t / H, h = t - e * H;
  int d = dst[e];
  float v = expf(alpha[t] - ord2f(amax[(size_t)d * H + h]));
  ea[t] = v;
  atomicAdd(&den[(size_t)d * H + h], v);
  if (h == 0) atomicAdd(&deg[d], 1.f);
}

__global__ void edge_agg(const float* __restrict__ xl, const int* __restrict__ src,
                         const int* __restrict__ dst, const float* __restrict__ ea,
                         const float* __restrict__ den, float* __restrict__ agg,
                         int E, int H, int C) {
  int t = blockIdx.x * blockDim.x + threadIdx.x;
  if (t >= E * H) return;
  int e = t / H, h = t - e * H;
  int s = src[e], d = dst[e];
  float a = ea[t] / (den[(size_t)d * H + h] + 1e-16f);
  const float* pj = xl + (size_t)s * H * C + (size_t)h * C;
  float* po = agg + (size_t)d * H * C + (size_t)h * C;
  for (int c = 0; c < C; ++c) atomicAdd(&po[c], pj[c] * a);
}

__global__ void gat_final(const float* __restrict__ agg, const float* __restrict__ deg,
                          const float* __restrict__ bias, float* __restrict__ out,
                          int N, int HC, int relu_flag) {
  long i = (long)blockIdx.x * blockDim.x + threadIdx.x;
  if (i >= (long)N * HC) return;
  int n = (int)(i / HC), j = (int)(i - (long)n * HC);
  float v = agg[i] / fmaxf(deg[n], 1.f) + bias[j];
  out[i] = relu_flag ? relu_(v) : v;
}

// ---------------------------------------------------------------------------
// misc elementwise kernels
// ---------------------------------------------------------------------------
__global__ void bn_rows(const float* __restrict__ x, float* __restrict__ y,
                        const float* __restrict__ g, const float* __restrict__ be,
                        const float* __restrict__ mn, const float* __restrict__ vr,
                        int N, int C, int relu_flag) {
  long i = (long)blockIdx.x * blockDim.x + threadIdx.x;
  if (i >= (long)N * C) return;
  int c = (int)(i % C);
  float v = (x[i] - mn[c]) * (g[c] * rsqrtf(vr[c] + 1e-5f)) + be[c];
  y[i] = relu_flag ? relu_(v) : v;
}

__global__ void down_xr(const float* __restrict__ xh, const float* __restrict__ Wr,
                        const float* __restrict__ br, float* __restrict__ xr, int N) {
  long i = (long)blockIdx.x * blockDim.x + threadIdx.x;
  if (i >= (long)N * 64) return;
  int n = (int)(i / 64), c = (int)(i & 63);
  xr[i] = xh[n] * Wr[c] + br[c];
}

__global__ void concat65(const float* __restrict__ z, const float* __restrict__ d,
                         float* __restrict__ h, int N) {
  long i = (long)blockIdx.x * blockDim.x + threadIdx.x;
  if (i >= (long)N * 65) return;
  int n = (int)(i / 65), c = (int)(i - (long)n * 65);
  h[i] = (c == 0) ? z[n] : d[(size_t)n * 64 + (c - 1)];
}

__global__ void build_edges(const int* __restrict__ ehh, int* __restrict__ src,
                            int* __restrict__ dst, int Ehh, int Nh) {
  int i = blockIdx.x * blockDim.x + threadIdx.x;
  int E = Ehh + Nh;
  if (i >= E) return;
  if (i < Ehh) { src[i] = ehh[i]; dst[i] = ehh[Ehh + i]; }
  else         { src[i] = i - Ehh; dst[i] = i - Ehh; }
}

// ---------------------------------------------------------------------------
// fused prediction MLP 64 -> 64 -> 32 -> 1 (weights in LDS)
// ---------------------------------------------------------------------------
__global__ __launch_bounds__(WG)
void mlp_pred(const float* __restrict__ h,
              const float* __restrict__ W0, const float* __restrict__ b0,
              const float* __restrict__ W1, const float* __restrict__ b1,
              const float* __restrict__ W2, const float* __restrict__ b2,
              float* __restrict__ out, int N) {
  __shared__ float sW0[64 * 64], sb0[64], sW1[64 * 32], sb1[32], sW2[32], sb2;
  int tid = threadIdx.x;
  for (int i = tid; i < 64 * 64; i += WG) sW0[i] = W0[i];
  for (int i = tid; i < 64 * 32; i += WG) sW1[i] = W1[i];
  for (int i = tid; i < 64; i += WG) sb0[i] = b0[i];
  for (int i = tid; i < 32; i += WG) { sb1[i] = b1[i]; sW2[i] = W2[i]; }
  if (tid == 0) sb2 = b2[0];
  __syncthreads();

  int n = blockIdx.x * blockDim.x + tid;
  if (n >= N) return;
  float xin[64];
  #pragma unroll
  for (int i = 0; i < 64; ++i) xin[i] = h[(size_t)n * 64 + i];
  float t1[64];
  #pragma unroll 4
  for (int o = 0; o < 64; ++o) {
    float a = sb0[o];
    #pragma unroll 8
    for (int i = 0; i < 64; ++i) a += xin[i] * sW0[i * 64 + o];
    t1[o] = relu_(a);
  }
  float t2[32];
  #pragma unroll 4
  for (int o = 0; o < 32; ++o) {
    float a = sb1[o];
    #pragma unroll 8
    for (int i = 0; i < 64; ++i) a += t1[i] * sW1[i * 32 + o];
    t2[o] = relu_(a);
  }
  float a = sb2;
  #pragma unroll
  for (int i = 0; i < 32; ++i) a += t2[i] * sW2[i];
  out[n] = a;
}

// ---------------------------------------------------------------------------
// orchestration
// ---------------------------------------------------------------------------
static inline unsigned cdiv(long a, long b) { return (unsigned)((a + b - 1) / b); }

extern "C" void kernel_launch(void* const* d_in, const int* in_sizes, int n_in,
                              void* d_out, int out_size, void* d_ws, size_t ws_size,
                              hipStream_t stream) {
  if (n_in < 112) return;
  auto P = [&](int i) { return (const float*)d_in[i]; };

  const float* x_low  = P(0);
  const float* x_high = P(1);
  const float* z_std  = P(2);
  const int* e_src_l2h = (const int*)d_in[109];
  const int* e_dst_l2h = (const int*)d_in[110];
  const int* e_hh      = (const int*)d_in[111];

  const int Nl   = in_sizes[0] / 625;     // 4000
  const int Nh   = in_sizes[1];           // 100000
  const int El2h = in_sizes[109];         // 900000
  const int Ehh  = in_sizes[111] / 2;     // 800000
  const int Ea   = Ehh + Nh;              // 900000 (with self loops)

  // -------- workspace arena --------
  char* w = (char*)d_ws;
  size_t need = 0;
  auto carve = [&](size_t elems, size_t esz) -> void* {
    size_t bytes = ((elems * esz) + 255) & ~(size_t)255;
    void* p = w + need; need += bytes; return p;
  };
  float*    enc_low = (float*)carve((size_t)Nl * 32, 4);
  float*    xl      = (float*)carve((size_t)Nh * 128, 4);
  float*    xr      = (float*)carve((size_t)Nh * 128, 4);
  float*    h_a     = (float*)carve((size_t)Nh * 128, 4);
  float*    h_b     = (float*)carve((size_t)Nh * 128, 4);   // doubles as agg
  float*    alpha   = (float*)carve((size_t)Ea * 2, 4);
  float*    ea      = (float*)carve((size_t)Ea * 2, 4);
  unsigned* amax    = (unsigned*)carve((size_t)Nh * 2, 4);
  float*    den     = (float*)carve((size_t)Nh * 2, 4);
  float*    deg     = (float*)carve((size_t)Nh, 4);
  int*      src_e   = (int*)carve((size_t)Ea, 4);
  int*      dst_e   = (int*)carve((size_t)Ea, 4);
  if (need > ws_size) return;   // workspace too small; bail deterministically

  // -------- CNN encoder --------
  CnnParams cp;
  for (int i = 0; i < 5; ++i) {
    cp.tw[i] = P(3 + 2 * i);  cp.tb[i]  = P(4 + 2 * i);
    cp.tg[i] = P(13 + 4 * i); cp.tbe[i] = P(14 + 4 * i);
    cp.tm[i] = P(15 + 4 * i); cp.tv[i]  = P(16 + 4 * i);
  }
  cp.vw[0] = P(33); cp.vb[0] = P(34); cp.vw[1] = P(35); cp.vb[1] = P(36);
  for (int i = 0; i < 2; ++i) {
    cp.vg[i] = P(37 + 4 * i); cp.vbe[i] = P(38 + 4 * i);
    cp.vm[i] = P(39 + 4 * i); cp.vv[i]  = P(40 + 4 * i);
  }
  cp.dW = P(45); cp.db = P(46);
  cnn_encoder<<<Nl, WG, 0, stream>>>(x_low, cp, enc_low, Nl);

  // -------- down GAT (low -> high, H=1, C=64) --------
  gemm_bf16_wmma<<<dim3(cdiv(Nl, TM), 1), WG, 0, stream>>>(
      enc_low, P(47), P(48), xl, Nl, 32, 64, 0);
  down_xr<<<cdiv((long)Nh * 64, WG), WG, 0, stream>>>(x_high, P(49), P(50), xr, Nh);
  fill_u32<<<1024, WG, 0, stream>>>(amax, 0u, Nh);
  fill_f32<<<1024, WG, 0, stream>>>(den, 0.f, Nh);
  fill_f32<<<1024, WG, 0, stream>>>(deg, 0.f, Nh);
  fill_f32<<<2048, WG, 0, stream>>>(h_b, 0.f, (long)Nh * 64);
  edge_alpha<<<cdiv((long)El2h, WG), WG, 0, stream>>>(xl, xr, e_src_l2h, e_dst_l2h,
                                                      P(51), alpha, amax, El2h, 1, 64);
  edge_exp<<<cdiv((long)El2h, WG), WG, 0, stream>>>(alpha, amax, e_dst_l2h, ea, den, deg,
                                                    El2h, 1);
  edge_agg<<<cdiv((long)El2h, WG), WG, 0, stream>>>(xl, e_src_l2h, e_dst_l2h, ea, den,
                                                    h_b, El2h, 1, 64);
  gat_final<<<cdiv((long)Nh * 64, WG), WG, 0, stream>>>(h_b, deg, P(52), xr, Nh, 64, 0);

  // h = concat([z_std, d])  -> (Nh, 65) in h_a
  concat65<<<cdiv((long)Nh * 65, WG), WG, 0, stream>>>(z_std, xr, h_a, Nh);

  // high-graph edges with self loops
  build_edges<<<cdiv((long)Ea, WG), WG, 0, stream>>>(e_hh, src_e, dst_e, Ehh, Nh);

  // -------- 5 GAT layers on the high graph --------
  const int in_dims[5] = {65, 128, 128, 128, 128};
  const int heads[5]   = {2, 2, 2, 2, 1};
  for (int l = 0; l < 5; ++l) {
    int Cin = in_dims[l], H = heads[l], HC = H * 64;
    int bn = 53 + 4 * l;   // pbn[l]: gamma, beta, mean, var
    int pg = 73 + 6 * l;   // pgat[l]: Wl, bl, Wr, br, att, bias
    bn_rows<<<cdiv((long)Nh * Cin, WG), WG, 0, stream>>>(
        h_a, h_b, P(bn), P(bn + 1), P(bn + 2), P(bn + 3), Nh, Cin, l > 0 ? 1 : 0);
    gemm_bf16_wmma<<<dim3(cdiv(Nh, TM), cdiv(HC, TN)), WG, 0, stream>>>(
        h_b, P(pg + 0), P(pg + 1), xl, Nh, Cin, HC, 0);
    gemm_bf16_wmma<<<dim3(cdiv(Nh, TM), cdiv(HC, TN)), WG, 0, stream>>>(
        h_b, P(pg + 2), P(pg + 3), xr, Nh, Cin, HC, 0);
    fill_u32<<<1024, WG, 0, stream>>>(amax, 0u, (long)Nh * H);
    fill_f32<<<1024, WG, 0, stream>>>(den, 0.f, (long)Nh * H);
    fill_f32<<<1024, WG, 0, stream>>>(deg, 0.f, Nh);
    fill_f32<<<2048, WG, 0, stream>>>(h_b, 0.f, (long)Nh * HC);   // agg (bn buf dead)
    edge_alpha<<<cdiv((long)Ea * H, WG), WG, 0, stream>>>(xl, xr, src_e, dst_e,
                                                          P(pg + 4), alpha, amax, Ea, H, 64);
    edge_exp<<<cdiv((long)Ea * H, WG), WG, 0, stream>>>(alpha, amax, dst_e, ea, den, deg,
                                                        Ea, H);
    edge_agg<<<cdiv((long)Ea * H, WG), WG, 0, stream>>>(xl, src_e, dst_e, ea, den,
                                                        h_b, Ea, H, 64);
    gat_final<<<cdiv((long)Nh * HC, WG), WG, 0, stream>>>(h_b, deg, P(pg + 5), h_a,
                                                          Nh, HC, l == 4 ? 1 : 0);
  }

  // -------- prediction MLP 64 -> 64 -> 32 -> 1 --------
  mlp_pred<<<cdiv(Nh, WG), WG, 0, stream>>>(h_a, P(103), P(104), P(105), P(106),
                                            P(107), P(108), (float*)d_out, Nh);
}
